// SMNet_36653250904097
// MI455X (gfx1250) — matrix operand order repaired and checked by reading, not compiled, not run on hardware
//
#include <hip/hip_runtime.h>
#include <hip/hip_bf16.h>

typedef _Float16 v8h  __attribute__((ext_vector_type(8)));
typedef _Float16 v16h __attribute__((ext_vector_type(16)));
typedef float    v8f  __attribute__((ext_vector_type(8)));
typedef float    v4f  __attribute__((ext_vector_type(4)));
typedef int      i4v  __attribute__((vector_size(16)));   // b128 payload

#define N_   2
#define T_   4
#define HW_  65536
#define EGO_ 64
#define MEM_ 128

#if __has_builtin(__builtin_amdgcn_global_load_async_to_lds_b128) && \
    __has_builtin(__builtin_amdgcn_s_wait_asynccnt)
#define USE_ASYNC_LDS 1
typedef __attribute__((address_space(1))) i4v* gas1_b128;
typedef __attribute__((address_space(3))) i4v* las3_b128;
#else
#define USE_ASYNC_LDS 0
#endif

// ---------------------------------------------------------------------------
// WMMA helpers (CDNA5 gfx1250, wave32)
// ---------------------------------------------------------------------------
__device__ __forceinline__ v8f wmma_f16(v16h a, v16h b, v8f c) {
  // (neg_a, A, neg_b, B, c_mod, C, reuse_a, reuse_b)
  return __builtin_amdgcn_wmma_f32_16x16x32_f16(false, a, false, b, (short)0, c,
                                                false, false);
}

// 16 consecutive halves -> B fragment slice for this lane (two 16B loads)
__device__ __forceinline__ v16h load16h(const _Float16* p) {
  v8h lo = *(const v8h*)p;
  v8h hi = *(const v8h*)(p + 8);
  v16h r;
#pragma unroll
  for (int i = 0; i < 8; ++i) { r[i] = lo[i]; r[8 + i] = hi[i]; }
  return r;
}

// A fragment slice: 8 halves at p, 8 halves at p+16 (ISA 16-bit A layout)
__device__ __forceinline__ v16h loadA16h(const _Float16* p) {
  v8h lo = *(const v8h*)p;
  v8h hi = *(const v8h*)(p + 16);
  v16h r;
#pragma unroll
  for (int i = 0; i < 8; ++i) { r[i] = lo[i]; r[8 + i] = hi[i]; }
  return r;
}

// ---------------------------------------------------------------------------
// Small utility kernels
// ---------------------------------------------------------------------------
__global__ void cvt_f32_f16_kernel(const float* __restrict__ src,
                                   _Float16* __restrict__ dst, long n) {
  for (long i = blockIdx.x * (long)blockDim.x + threadIdx.x; i < n;
       i += (long)gridDim.x * blockDim.x)
    dst[i] = (_Float16)src[i];
}

// f32 OIHW -> f16 [kh*KW+kw][coutp][cinp], zero-padded both cin and cout
__global__ void wtrans_kernel(const float* __restrict__ w,
                              _Float16* __restrict__ wt, int Cout, int Coutp,
                              int Cin, int Cinp, int KH, int KW) {
  long total = (long)KH * KW * Coutp * Cinp;
  for (long i = blockIdx.x * (long)blockDim.x + threadIdx.x; i < total;
       i += (long)gridDim.x * blockDim.x) {
    int ci = (int)(i % Cinp);
    long r = i / Cinp;
    int co = (int)(r % Coutp);
    int tap = (int)(r / Coutp);
    int kh = tap / KW, kw = tap % KW;
    _Float16 v = (_Float16)0.0f;
    if (ci < Cin && co < Cout)
      v = (_Float16)w[(((long)co * Cin + ci) * KH + kh) * KW + kw];
    wt[i] = v;
  }
}

__global__ void zero_flags_kernel(int* flags) {
  if (threadIdx.x < 8 && blockIdx.x == 0) flags[threadIdx.x] = 0;
}

__global__ void any_kernel(const int* __restrict__ inl, int* flags) {
  long total = (long)N_ * T_ * HW_;
  for (long i = blockIdx.x * (long)blockDim.x + threadIdx.x; i < total;
       i += (long)gridDim.x * blockDim.x) {
    if (inl[i] > 0) {
      int nt = (int)(i >> 16);
      if (flags[nt] == 0) atomicOr(&flags[nt], 1);
    }
  }
}

__global__ void obs_kernel(const int* __restrict__ inl,
                           const int* __restrict__ flags,
                           float* __restrict__ out) {
  long total = (long)N_ * HW_;
  for (long i = blockIdx.x * (long)blockDim.x + threadIdx.x; i < total;
       i += (long)gridDim.x * blockDim.x) {
    int n = (int)(i >> 16);
    int p = (int)(i & (HW_ - 1));
    int acc = 0;
#pragma unroll
    for (int t = 0; t < T_; ++t)
      acc += inl[((long)(n * T_ + t) << 16) + p] * flags[n * T_ + t];
    out[i] = (float)acc;
  }
}

// ---------------------------------------------------------------------------
// GRU over pixels: WMMA GEMM, M=16 pixels per wave, N=16 gate units per tile
// ---------------------------------------------------------------------------
__global__ __launch_bounds__(256) void gru_kernel(
    const float* __restrict__ feat,     // [N,T,256,256,64]
    const int* __restrict__ inl,        // [N,T,256,256]
    const _Float16* __restrict__ wih,   // [384][64]  f16
    const _Float16* __restrict__ whh,   // [384][128] f16
    const float* __restrict__ bih, const float* __restrict__ bhh,
    _Float16* __restrict__ act0) {      // [N,256,256,128] NHWC f16
  __shared__ _Float16 hbuf[8][16 * MEM_];
  __shared__ float mrowS[8][16];

  const int tid = threadIdx.x;
  const int wv = tid >> 5;
  const int l = tid & 31;
  const int lm = l & 15;
  const int lh = l >> 4;

  const long gp = (long)blockIdx.x * 128 + wv * 16;  // base pixel over N*HW
  const int n = (int)(gp >> 16);
  const int p = (int)(gp & (HW_ - 1));

  // zero hidden state tile
  for (int i = l; i < 16 * MEM_ / 8; i += 32) {
    v8h z = {};
    *(v8h*)&hbuf[wv][i * 8] = z;
  }
  __builtin_amdgcn_wave_barrier();

#pragma unroll 1
  for (int t = 0; t < T_; ++t) {
    // ---- A fragments for x: [16 pixels x 64 ch], f32 -> f16
    const float* fb = feat + (((long)(n * T_ + t) * HW_) + p) * EGO_;
    v16h xa[2];
#pragma unroll
    for (int f = 0; f < 2; ++f) {
      const float* src = fb + (long)lm * EGO_;
      int k0 = f * 32 + lh * 8;
      v4f u0 = *(const v4f*)(src + k0);
      v4f u1 = *(const v4f*)(src + k0 + 4);
      v4f u2 = *(const v4f*)(src + k0 + 16);
      v4f u3 = *(const v4f*)(src + k0 + 20);
      v16h a;
#pragma unroll
      for (int i = 0; i < 4; ++i) {
        a[i] = (_Float16)u0[i];      a[4 + i] = (_Float16)u1[i];
        a[8 + i] = (_Float16)u2[i];  a[12 + i] = (_Float16)u3[i];
      }
      xa[f] = a;
    }

    // ---- per-pixel inlier mask for this wave
    const int* mb = inl + ((long)(n * T_ + t) * HW_) + p;
    if (l < 16) mrowS[wv][l] = (mb[l] > 0) ? 1.0f : 0.0f;
    __builtin_amdgcn_wave_barrier();
    float mr[8];
#pragma unroll
    for (int r = 0; r < 8; ++r) mr[r] = mrowS[wv][r + lh * 8];

    // ---- A fragments for h from LDS
    v16h ha[4];
#pragma unroll
    for (int f = 0; f < 4; ++f)
      ha[f] = loadA16h(&hbuf[wv][lm * MEM_ + f * 32 + lh * 8]);

    // ---- 8 output tiles of 16 hidden units each.
    // Keep this loop rolled: one tile's accumulators + fragments live at a
    // time (full unroll spilled B fragments to scratch in a prior build).
#pragma unroll 1
    for (int jt = 0; jt < 8; ++jt) {
      const int j = jt * 16 + lm;
      // per-lane gate biases loaded in-loop (L1-hot; avoids movrel arrays)
      const float bR = bih[j] + bhh[j];
      const float bZ = bih[128 + j] + bhh[128 + j];
      const float bNi = bih[256 + j];
      const float bNh = bhh[256 + j];

      v8f aR = {}, aZ = {}, aNi = {}, aNh = {};
      const int row = jt * 16 + lm;
      const int bo = lh * 8;
#pragma unroll
      for (int ks = 0; ks < 2; ++ks) {  // K over x channels (64)
        int ko = ks * 32 + bo;
        aR  = wmma_f16(xa[ks], loadA16h(wih + (long)(row)*EGO_ + ko), aR);
        aZ  = wmma_f16(xa[ks], loadA16h(wih + (long)(128 + row) * EGO_ + ko), aZ);
        aNi = wmma_f16(xa[ks], loadA16h(wih + (long)(256 + row) * EGO_ + ko), aNi);
      }
#pragma unroll
      for (int ks = 0; ks < 4; ++ks) {  // K over h channels (128)
        int ko = ks * 32 + bo;
        aR  = wmma_f16(ha[ks], loadA16h(whh + (long)(row)*MEM_ + ko), aR);
        aZ  = wmma_f16(ha[ks], loadA16h(whh + (long)(128 + row) * MEM_ + ko), aZ);
        aNh = wmma_f16(ha[ks], loadA16h(whh + (long)(256 + row) * MEM_ + ko), aNh);
      }
      // gate math + masked state update (C layout: vgpr r -> pixel, lane -> unit)
#pragma unroll
      for (int r = 0; r < 8; ++r) {
        float rg = 1.0f / (1.0f + __expf(-(aR[r] + bR)));
        float zg = 1.0f / (1.0f + __expf(-(aZ[r] + bZ)));
        float ng = tanhf(aNi[r] + bNi + rg * (aNh[r] + bNh));
        int pix = r + lh * 8;
        int off = pix * MEM_ + jt * 16 + lm;
        float hold = (float)hbuf[wv][off];
        float hnew = (1.0f - zg) * ng + zg * hold;
        float hsel = mr[r] * hnew + (1.0f - mr[r]) * hold;
        hbuf[wv][off] = (_Float16)hsel;
      }
    }
    __builtin_amdgcn_wave_barrier();
  }

  // ---- write final state NHWC f16
  _Float16* ob = act0 + (((long)n * HW_) + p) * MEM_;
#pragma unroll
  for (int i = 0; i < 8; ++i) {
    int c = (lh * 8 + i) * 8;
    *(v8h*)(ob + (long)lm * MEM_ + c) = *(const v8h*)&hbuf[wv][lm * MEM_ + c];
  }
}

// ---------------------------------------------------------------------------
// Implicit-GEMM conv with WMMA.
// Block = 32-pixel row strip. 8 waves = 2 pixel sub-strips x 4 cout-pair cols.
// Each wave computes 16 pixels x 32 couts (one A fragment feeds two WMMAs).
// in: NHWC f16 (Cinp padded), wt: [tap][Coutp][Cinp] f16, out: NHWC f32
// ---------------------------------------------------------------------------
__global__ __launch_bounds__(256) void conv_wmma_kernel(
    const _Float16* __restrict__ in, const _Float16* __restrict__ wt,
    float* __restrict__ out, int Cinp, int Cout, int Coutp, int KH, int KW,
    int pad) {
  extern __shared__ _Float16 tile[];  // [KH][32+KW-1][Cinp]
  const int tid = threadIdx.x;
  const int wv = tid >> 5;
  const int l = tid & 31;
  const int lm = l & 15;
  const int lh = l >> 4;

  const int bx = blockIdx.x;
  const int xbase = (bx & 7) * 32;
  const int y = (bx >> 3) & 255;
  const int n = bx >> 11;
  const int TW = 32 + KW - 1;

  // ---- stage input tile (with halo + zero padding) into LDS
  const int c8n = Cinp >> 3;
  const int cnt8 = KH * TW * c8n;
  for (int i = tid; i < cnt8; i += 256) {
    int rc = i / c8n;
    int ci8 = i - rc * c8n;
    int col = rc % TW, row = rc / TW;
    int yy = y + row - pad;
    int xx = xbase + col - pad;
    _Float16* lp = &tile[(long)i * 8];
    if (yy >= 0 && yy < 256 && xx >= 0 && xx < 256) {
      const _Float16* gp =
          in + (((long)n * 256 + yy) * 256 + xx) * Cinp + ci8 * 8;
#if USE_ASYNC_LDS
      __builtin_amdgcn_global_load_async_to_lds_b128((gas1_b128)gp,
                                                     (las3_b128)lp, 0, 0);
#else
      *(v8h*)lp = *(const v8h*)gp;
#endif
    } else {
      v8h z = {};
      *(v8h*)lp = z;
    }
  }
#if USE_ASYNC_LDS
  __builtin_amdgcn_s_wait_asynccnt(0);  // each wave drains its own copies
#endif
  __syncthreads();

  const int sub = wv >> 2;         // which 16-pixel half of the strip
  const int jb = (wv & 3) * 32;    // cout pair base
  if (jb < Coutp) {  // wave-uniform: EXEC stays all-ones for WMMA
    v8f acc0 = {}, acc1 = {};
#pragma unroll 1
    for (int kh = 0; kh < KH; ++kh)
#pragma unroll 1
      for (int kw = 0; kw < KW; ++kw) {
        const _Float16* wb0 =
            wt + ((long)(kh * KW + kw) * Coutp + jb + lm) * Cinp + lh * 8;
        const _Float16* wb1 = wb0 + (long)16 * Cinp;
        const int abase = (kh * TW + sub * 16 + lm + kw) * Cinp + lh * 8;
        __builtin_prefetch(wb0 + (long)Coutp * Cinp, 0, 1);  // next tap
        for (int kc = 0; kc < Cinp; kc += 32) {
          v16h a = loadA16h(&tile[abase + kc]);
          acc0 = wmma_f16(a, loadA16h(wb0 + kc), acc0);
          acc1 = wmma_f16(a, loadA16h(wb1 + kc), acc1);
        }
      }
    // ---- store raw conv output, NHWC f32 (lane -> cout: coalesced)
    float* ob =
        out + (((long)n * 256 + y) * 256 + xbase + sub * 16) * Cout + jb + lm;
    if (jb < Cout) {
#pragma unroll
      for (int r = 0; r < 8; ++r) {
        int pix = r + lh * 8;
        ob[(long)pix * Cout] = acc0[r];
      }
    }
    if (jb + 16 < Cout) {
#pragma unroll
      for (int r = 0; r < 8; ++r) {
        int pix = r + lh * 8;
        ob[(long)pix * Cout + 16] = acc1[r];
      }
    }
  }
}

// ---------------------------------------------------------------------------
// Deterministic per-channel batch stats -> fused scale/shift
// ---------------------------------------------------------------------------
__global__ __launch_bounds__(256) void bn_stats_kernel(
    const float* __restrict__ raw, int C, int count,
    const float* __restrict__ g, const float* __restrict__ b,
    float* __restrict__ sa, float* __restrict__ sb) {
  __shared__ float s1[256], s2[256];
  const int c = blockIdx.x;
  const int tid = threadIdx.x;
  float sum = 0.f, sq = 0.f;
  for (long i = tid; i < count; i += 256) {
    float v = raw[i * (long)C + c];
    sum += v;
    sq += v * v;
  }
  s1[tid] = sum;
  s2[tid] = sq;
  __syncthreads();
  for (int s = 128; s > 0; s >>= 1) {
    if (tid < s) { s1[tid] += s1[tid + s]; s2[tid] += s2[tid + s]; }
    __syncthreads();
  }
  if (tid == 0) {
    float inv = 1.0f / (float)count;
    float mu = s1[0] * inv;
    float var = s2[0] * inv - mu * mu;
    float rs = rsqrtf(var + 1e-5f);
    sa[c] = g[c] * rs;
    sb[c] = b[c] - mu * g[c] * rs;
  }
}

// normalize + ReLU + f16 convert, pad channels to Cp with zeros
__global__ void bn_apply_kernel(const float* __restrict__ raw, int C, int Cp,
                                const float* __restrict__ sa,
                                const float* __restrict__ sb,
                                _Float16* __restrict__ act, long total) {
  for (long i = blockIdx.x * (long)blockDim.x + threadIdx.x; i < total;
       i += (long)gridDim.x * blockDim.x) {
    int c = (int)(i % Cp);
    long pix = i / Cp;
    float v = 0.0f;
    if (c < C) {
      v = raw[pix * (long)C + c] * sa[c] + sb[c];
      v = v > 0.0f ? v : 0.0f;
    }
    act[i] = (_Float16)v;
  }
}

// final 1x1 conv 48->20 (tiny) : NCHW f32 output into d_out
__global__ void obj2_kernel(const _Float16* __restrict__ act,  // [N,HW,64]
                            const float* __restrict__ w,       // [20][48]
                            const float* __restrict__ b,
                            float* __restrict__ out) {         // [N,20,HW]
  long total = (long)N_ * 20 * HW_;
  for (long i = blockIdx.x * (long)blockDim.x + threadIdx.x; i < total;
       i += (long)gridDim.x * blockDim.x) {
    int p = (int)(i & (HW_ - 1));
    long r = i >> 16;
    int o = (int)(r % 20);
    int n = (int)(r / 20);
    const _Float16* ap = act + (((long)n << 16) + p) * 64;
    const float* wr = w + o * 48;
    float s = b[o];
#pragma unroll 8
    for (int c = 0; c < 48; ++c) s += (float)ap[c] * wr[c];
    out[i] = s;
  }
}

// ---------------------------------------------------------------------------
// Host-side orchestration
// ---------------------------------------------------------------------------
static inline size_t alignup(size_t x) { return (x + 255) & ~(size_t)255; }

extern "C" void kernel_launch(void* const* d_in, const int* in_sizes, int n_in,
                              void* d_out, int out_size, void* d_ws,
                              size_t ws_size, hipStream_t stream) {
  const float* features = (const float*)d_in[0];
  const int* masks = (const int*)d_in[1];
  const float* w_ih = (const float*)d_in[2];
  const float* w_hh = (const float*)d_in[3];
  const float* b_ih = (const float*)d_in[4];
  const float* b_hh = (const float*)d_in[5];
  const float* conv1_w = (const float*)d_in[6];
  const float* bn1_g = (const float*)d_in[7];
  const float* bn1_b = (const float*)d_in[8];
  const float* conv2_w = (const float*)d_in[9];
  const float* bn2_g = (const float*)d_in[10];
  const float* bn2_b = (const float*)d_in[11];
  const float* conv3_w = (const float*)d_in[12];
  const float* bn3_g = (const float*)d_in[13];
  const float* bn3_b = (const float*)d_in[14];
  const float* obj1_w = (const float*)d_in[15];
  const float* bn4_g = (const float*)d_in[16];
  const float* bn4_b = (const float*)d_in[17];
  const float* obj2_w = (const float*)d_in[18];
  const float* obj2_b = (const float*)d_in[19];

  float* semmap = (float*)d_out;                         // [2,20,256,256]
  float* obs_out = (float*)d_out + (long)N_ * 20 * HW_;  // [2,256,256]

  // workspace carve-up
  char* ws = (char*)d_ws;
  size_t off = 0;
  int* flags = (int*)(ws + off);          off = alignup(off + 32);
  _Float16* wihH = (_Float16*)(ws + off); off = alignup(off + 384 * 64 * 2);
  _Float16* whhH = (_Float16*)(ws + off); off = alignup(off + 384 * 128 * 2);
  _Float16* wt1 = (_Float16*)(ws + off);  off = alignup(off + (size_t)49 * 128 * 128 * 2);
  _Float16* wt2 = (_Float16*)(ws + off);  off = alignup(off + (size_t)9 * 64 * 128 * 2);
  _Float16* wt3 = (_Float16*)(ws + off);  off = alignup(off + (size_t)9 * 64 * 64 * 2);
  _Float16* wt4 = (_Float16*)(ws + off);  off = alignup(off + (size_t)9 * 64 * 64 * 2);
  float* sa = (float*)(ws + off);         off = alignup(off + 128 * 4);
  float* sb = (float*)(ws + off);         off = alignup(off + 128 * 4);
  _Float16* actA = (_Float16*)(ws + off); off = alignup(off + (size_t)N_ * HW_ * 128 * 2);
  _Float16* actB = (_Float16*)(ws + off); off = alignup(off + (size_t)N_ * HW_ * 128 * 2);
  float* raw = (float*)(ws + off);        off = alignup(off + (size_t)N_ * HW_ * 128 * 4);
  (void)ws_size;

  const int TPB = 256;
  const int count = N_ * HW_;  // per-channel BN population

  // ---- observed_masks path
  zero_flags_kernel<<<1, 32, 0, stream>>>(flags);
  any_kernel<<<1024, TPB, 0, stream>>>(masks, flags);
  obs_kernel<<<512, TPB, 0, stream>>>(masks, flags, obs_out);

  // ---- GRU weights to f16, then GRU
  cvt_f32_f16_kernel<<<96, TPB, 0, stream>>>(w_ih, wihH, 384L * 64);
  cvt_f32_f16_kernel<<<192, TPB, 0, stream>>>(w_hh, whhH, 384L * 128);
  gru_kernel<<<1024, TPB, 0, stream>>>(features, masks, wihH, whhH, b_ih, b_hh,
                                       actA);

  // ---- weight transposes for decoder convs (cout padded to 32-multiples)
  wtrans_kernel<<<3136, TPB, 0, stream>>>(conv1_w, wt1, 128, 128, 128, 128, 7, 7);
  wtrans_kernel<<<288, TPB, 0, stream>>>(conv2_w, wt2, 64, 64, 128, 128, 3, 3);
  wtrans_kernel<<<144, TPB, 0, stream>>>(conv3_w, wt3, 48, 64, 64, 64, 3, 3);
  wtrans_kernel<<<144, TPB, 0, stream>>>(obj1_w, wt4, 48, 64, 64, 64, 3, 3);

  const int convGrid = N_ * 256 * 8;  // (n, y, 32-pixel x strip)

  // ---- layer 1: 7x7 128->128
  conv_wmma_kernel<<<convGrid, TPB, 7 * 38 * 128 * 2, stream>>>(
      actA, wt1, raw, 128, 128, 128, 7, 7, 3);
  bn_stats_kernel<<<128, TPB, 0, stream>>>(raw, 128, count, bn1_g, bn1_b, sa, sb);
  bn_apply_kernel<<<8192, TPB, 0, stream>>>(raw, 128, 128, sa, sb, actB,
                                            (long)count * 128);

  // ---- layer 2: 3x3 128->64
  conv_wmma_kernel<<<convGrid, TPB, 3 * 34 * 128 * 2, stream>>>(
      actB, wt2, raw, 128, 64, 64, 3, 3, 1);
  bn_stats_kernel<<<64, TPB, 0, stream>>>(raw, 64, count, bn2_g, bn2_b, sa, sb);
  bn_apply_kernel<<<8192, TPB, 0, stream>>>(raw, 64, 64, sa, sb, actA,
                                            (long)count * 64);

  // ---- layer 3: 3x3 64->48 (channels padded to 64 for next layer's K)
  conv_wmma_kernel<<<convGrid, TPB, 3 * 34 * 64 * 2, stream>>>(
      actA, wt3, raw, 64, 48, 64, 3, 3, 1);
  bn_stats_kernel<<<48, TPB, 0, stream>>>(raw, 48, count, bn3_g, bn3_b, sa, sb);
  bn_apply_kernel<<<8192, TPB, 0, stream>>>(raw, 48, 64, sa, sb, actB,
                                            (long)count * 64);

  // ---- layer 4 (obj1): 3x3 48->48
  conv_wmma_kernel<<<convGrid, TPB, 3 * 34 * 64 * 2, stream>>>(
      actB, wt4, raw, 64, 48, 64, 3, 3, 1);
  bn_stats_kernel<<<48, TPB, 0, stream>>>(raw, 48, count, bn4_g, bn4_b, sa, sb);
  bn_apply_kernel<<<8192, TPB, 0, stream>>>(raw, 48, 64, sa, sb, actA,
                                            (long)count * 64);

  // ---- obj2: 1x1 48->20 + bias -> semmap (NCHW f32)
  obj2_kernel<<<4096, TPB, 0, stream>>>(actA, obj2_w, obj2_b, semmap);
}
